// MultiHeadAttention_1795296329962
// MI455X (gfx1250) — matrix operand order repaired and checked by reading, not compiled, not run on hardware
//
#include <hip/hip_runtime.h>

// ---------------------------------------------------------------------------
// MHA forward for MI455X (gfx1250, wave32, WMMA, async-to-LDS, TDM).
//   B=2, S=2048, DM=1024, H=16, D=64.
// Bandwidth-bound: must write 512MB fp32 attn tensor (~22us @ 23.3 TB/s).
// bf16 v_wmma_f32_16x16x32_bf16 with fp32 accumulation keeps the ~69 GFLOP
// of matrix math below that roofline. Softmax is exact (two-pass online).
// Data movement: TDM (tensor_load_to_lds) for contiguous/strided bf16 tiles,
// global_load_async_to_lds_b128 for per-lane copies, native v_cvt bf16.
// ---------------------------------------------------------------------------

#define DM_ 1024
#define S_  2048
#define B_  2
#define H_  16
#define D_  64
#define NT_ (B_ * S_)   // 4096 tokens
#define KC_ 128         // K/V streaming chunk (tokens)

typedef __attribute__((ext_vector_type(16))) __bf16          v16bf;
typedef __attribute__((ext_vector_type(2)))  __bf16          v2bf;
typedef __attribute__((ext_vector_type(16))) unsigned short  v16us;
typedef __attribute__((ext_vector_type(8)))  float           v8f;
typedef __attribute__((ext_vector_type(4)))  unsigned int    v4u;
typedef __attribute__((ext_vector_type(8)))  int             v8i;
typedef __attribute__((ext_vector_type(4)))  int             v4i;

// float -> bf16 via native hardware conversion (v_cvt*_bf16_f32)
__device__ __forceinline__ unsigned short f2bf(float f) {
  return __builtin_bit_cast(unsigned short, (__bf16)f);
}
__device__ __forceinline__ unsigned int pack2(float a, float b) {
  v2bf v = {(__bf16)a, (__bf16)b};
  return __builtin_bit_cast(unsigned int, v);
}

__device__ __forceinline__ v8f wmma_bf16(v16us a, v16us b, v8f c) {
  // (neg_a, A, neg_b, B, c_mod, C, reuse_a, reuse_b)
  return __builtin_amdgcn_wmma_f32_16x16x32_bf16(
      false, __builtin_bit_cast(v16bf, a),
      false, __builtin_bit_cast(v16bf, b),
      (short)0, c, false, false);
}

// LDS byte offset of a generic pointer to __shared__ (aperture lives in the
// high 32 bits on gfx1250; low 32 bits are the in-allocation byte address).
__device__ __forceinline__ unsigned lds_off(const void* p) {
  return (unsigned)(size_t)p;
}
// Per-lane async copy: LDS[lds] = MEM[g] (16 bytes). Tracked by ASYNCcnt.
__device__ __forceinline__ void async_b128(const void* g, unsigned lds) {
  asm volatile("global_load_async_to_lds_b128 %0, %1, off"
               :: "v"(lds), "v"(g) : "memory");
}
__device__ __forceinline__ void wait_async0() {
  asm volatile("s_wait_asynccnt 0" ::: "memory");
}

// Tensor Data Mover: DMA a 2D bf16 tile (tile_x elems/row, tile_y rows, row
// stride row_stride elems) from global to LDS (rows stored contiguously).
// D# layout per CDNA5 ISA 8.3/8.4: group0 = {count=1, lds_addr, global_addr,
// type=2}; group1 = {data_size=2B, tensor_dim0/1, tile_dim0/1, dim0_stride}.
// Issue from one wave only (TDM ignores EXEC; done signal via TENSORcnt).
// This toolchain's builtin is the 6-arg form: (g0, g1, g2, g3, g4, cpol).
__device__ __forceinline__ void tdm_load_2d_bf16(const void* gaddr, unsigned lds_addr,
                                                 unsigned tile_x, unsigned tile_y,
                                                 unsigned tensor_y, unsigned row_stride) {
  const unsigned long long ga = (unsigned long long)(size_t)gaddr;
  const v4u g0 = {1u,                                            // count=1
                  lds_addr,                                      // bits 63:32
                  (unsigned)ga,                                  // gaddr lo
                  ((unsigned)(ga >> 32) & 0x01ffffffu) | (2u << 30)};  // type=2
  const v8i g1 = {(int)(1u << 16),              // data_size code 1 (2 bytes)
                  (int)(tile_x << 16),          // tensor_dim0[15:0] @ 63:48
                  (int)(tensor_y << 16),        // tensor_dim1[15:0] @ 95:80
                  (int)(tile_x << 16),          // tile_dim0 @ 127:112
                  (int)tile_y,                  // tile_dim1 @ 143:128
                  (int)row_stride,              // tensor_dim0_stride lo32
                  0, 0};
  const v4i z4 = {0, 0, 0, 0};
  const v8i z8 = {0, 0, 0, 0, 0, 0, 0, 0};
  __builtin_amdgcn_tensor_load_to_lds(g0, g1, z4, z4, z8, 0);
}
__device__ __forceinline__ void wait_tensor0() {
  __builtin_amdgcn_s_wait_tensorcnt(0);
}

// A fragment: 16(M) x 32(K) bf16, row-major source, leading dim lda (elems).
__device__ __forceinline__ v16us frag_a(const unsigned short* p0, int lda, int lane) {
  const unsigned short* p = p0 + (size_t)(lane & 15) * lda + ((lane >> 4) & 1) * 16;
  v16us a;
#pragma unroll
  for (int i = 0; i < 8; ++i) {
    unsigned int w = *(const unsigned int*)(p + 2 * i);
    a[2 * i]     = (unsigned short)w;  a[2 * i + 1] = (unsigned short)(w >> 16);
  }
  return a;
}

// B fragment: 32(K) x 16(N) from N-major storage src[n*ldn + k] (K contiguous).
__device__ __forceinline__ v16us frag_b_cont(const unsigned short* src, int ldn,
                                             int ncol, int kbase, int lane) {
  const unsigned short* p =
      src + (size_t)(ncol + (lane & 15)) * ldn + kbase + ((lane >> 4) & 1) * 16;
  v16us b;
#pragma unroll
  for (int i = 0; i < 8; ++i) {
    unsigned int w = *(const unsigned int*)(p + 2 * i);
    b[2 * i] = (unsigned short)w;
    b[2 * i + 1] = (unsigned short)(w >> 16);
  }
  return b;
}

// ---------------------------------------------------------------------------
// Kernel 1: Y = X(fp32)*W(fp32) + bias, output bf16 head-split [B][H][S][D].
// Block = 256 (8 waves), tile 128(M) x 64(N), K-step 32; waves 4x2, 2x2 WMMA.
// ---------------------------------------------------------------------------
__global__ __launch_bounds__(256) void proj_gemm_kernel(
    const float* __restrict__ X, const float* __restrict__ W,
    const float* __restrict__ bias, unsigned short* __restrict__ dst) {
  __shared__ __align__(16) unsigned short As[128 * 32];  // row-major M x K
  __shared__ __align__(16) unsigned short Bt[64 * 32];   // N-major: [n][k]
  const int t = threadIdx.x, lane = t & 31, wave = t >> 5;
  const int wm = wave >> 1, wn = wave & 1;
  const int m0 = blockIdx.y * 128, n0 = blockIdx.x * 64;

  v8f acc[2][2];
#pragma unroll
  for (int i = 0; i < 2; ++i)
#pragma unroll
    for (int j = 0; j < 2; ++j)
#pragma unroll
      for (int r = 0; r < 8; ++r) acc[i][j][r] = 0.f;

  for (int k0 = 0; k0 < DM_; k0 += 32) {
    {  // stage X tile 128x32 (fp32 -> bf16 via v_cvt), float4 loads
      const int row = t >> 1, c0 = (t & 1) * 16;
      const float4* src = (const float4*)(X + (size_t)(m0 + row) * DM_ + k0 + c0);
      unsigned int* d = (unsigned int*)&As[row * 32 + c0];
#pragma unroll
      for (int i = 0; i < 4; ++i) {
        float4 f = src[i];
        d[2 * i] = pack2(f.x, f.y);
        d[2 * i + 1] = pack2(f.z, f.w);
      }
    }
    {  // stage W tile 32x64 transposed: thread owns k-pair (2pk,2pk+1), 4 n's
      const int pk = t >> 4, c = (t & 15) * 4;
      const float4 f0 = *(const float4*)(W + (size_t)(k0 + 2 * pk) * DM_ + n0 + c);
      const float4 f1 = *(const float4*)(W + (size_t)(k0 + 2 * pk + 1) * DM_ + n0 + c);
      *(unsigned int*)&Bt[(c + 0) * 32 + 2 * pk] = pack2(f0.x, f1.x);
      *(unsigned int*)&Bt[(c + 1) * 32 + 2 * pk] = pack2(f0.y, f1.y);
      *(unsigned int*)&Bt[(c + 2) * 32 + 2 * pk] = pack2(f0.z, f1.z);
      *(unsigned int*)&Bt[(c + 3) * 32 + 2 * pk] = pack2(f0.w, f1.w);
    }
    __syncthreads();
    const v16us a0 = frag_a(&As[(wm * 32 + 0) * 32], 32, lane);
    const v16us a1 = frag_a(&As[(wm * 32 + 16) * 32], 32, lane);
    const v16us b0 = frag_b_cont(Bt, 32, wn * 32 + 0, 0, lane);
    const v16us b1 = frag_b_cont(Bt, 32, wn * 32 + 16, 0, lane);
    acc[0][0] = wmma_bf16(a0, b0, acc[0][0]);
    acc[0][1] = wmma_bf16(a0, b1, acc[0][1]);
    acc[1][0] = wmma_bf16(a1, b0, acc[1][0]);
    acc[1][1] = wmma_bf16(a1, b1, acc[1][1]);
    __syncthreads();
  }

  const int nn = lane & 15, half = (lane >> 4) & 1;
#pragma unroll
  for (int i = 0; i < 2; ++i)
#pragma unroll
    for (int j = 0; j < 2; ++j) {
      const int ncol = n0 + wn * 32 + j * 16 + nn;
      const float bv = bias[ncol];
      const int hh = ncol >> 6, dd = ncol & 63;
#pragma unroll
      for (int r = 0; r < 8; ++r) {
        const int mrow = m0 + wm * 32 + i * 16 + r + 8 * half;
        const int bb = mrow >> 11, ss = mrow & (S_ - 1);
        dst[(((size_t)bb * H_ + hh) * S_ + ss) * D_ + dd] = f2bf(acc[i][j][r] + bv);
      }
    }
}

// ---------------------------------------------------------------------------
// Kernel 2: attention per (b,h). Each wave owns a 16-row Q tile; K/V stream
// through LDS in 128-token chunks. Pass-1 K arrives via TDM, pass-2 K via
// per-lane async-to-LDS; V is staged transposed (d-major, K-pair packed).
// ---------------------------------------------------------------------------
__global__ __launch_bounds__(256) void attn_kernel(
    const unsigned short* __restrict__ qh, const unsigned short* __restrict__ kh,
    const unsigned short* __restrict__ vh, const float* __restrict__ mask,
    float* __restrict__ attn, unsigned short* __restrict__ oh) {
  __shared__ __align__(16) unsigned short Ks[KC_ * D_];   // token-major, 16 KB
  __shared__ __align__(16) unsigned short Vt[D_ * KC_];   // d-major,     16 KB
  __shared__ __align__(16) unsigned short Ps[8][16 * 32]; // P re-layout,  8 KB
  __shared__ float Ms[KC_];                               // mask * -1e9

  const int t = threadIdx.x, lane = t & 31, wave = t >> 5;
  const int bh = blockIdx.y;          // b*H + h
  const int b = bh >> 4, h = bh & 15;
  const int q0 = blockIdx.x * 128 + wave * 16;
  const int nn = lane & 15, half = (lane >> 4) & 1;

  const unsigned short* Qb = qh + (size_t)bh * S_ * D_;
  const unsigned short* Kb = kh + (size_t)bh * S_ * D_;
  const unsigned short* Vb = vh + (size_t)bh * S_ * D_;

  // Q tile (16 x 64) fragments live in registers for the whole kernel.
  const v16us qa0 = frag_a(Qb + (size_t)q0 * D_ + 0, D_, lane);
  const v16us qa1 = frag_a(Qb + (size_t)q0 * D_ + 32, D_, lane);

  float mL[8], lL[8];
#pragma unroll
  for (int r = 0; r < 8; ++r) { mL[r] = -1e30f; lL[r] = 0.f; }

  // ---- pass 1: per-lane online (max, sumexp) over all 2048 keys ----
  for (int k0 = 0; k0 < S_; k0 += KC_) {
    if (t < 32)  // one TDM issue per workgroup: 128x64 bf16 tile -> Ks
      tdm_load_2d_bf16(Kb + (size_t)k0 * D_, lds_off(Ks), D_, KC_,
                       (unsigned)(S_ - k0), D_);
    if (t < KC_) Ms[t] = mask[(size_t)b * S_ + k0 + t] * (-1e9f);
    if (t < 32) wait_tensor0();
    __syncthreads();
#pragma unroll
    for (int kt = 0; kt < KC_ / 16; ++kt) {
      v8f acc;
#pragma unroll
      for (int r = 0; r < 8; ++r) acc[r] = 0.f;
      acc = wmma_bf16(qa0, frag_b_cont(Ks, D_, kt * 16, 0, lane), acc);
      acc = wmma_bf16(qa1, frag_b_cont(Ks, D_, kt * 16, 32, lane), acc);
      const float mk = Ms[kt * 16 + nn];
#pragma unroll
      for (int r = 0; r < 8; ++r) {
        const float val = acc[r] * 0.125f + mk;   // 1/sqrt(64)
        const float mo = mL[r];
        const float mx = fmaxf(mo, val);
        lL[r] = lL[r] * __expf(mo - mx) + __expf(val - mx);
        mL[r] = mx;
      }
    }
    __syncthreads();
  }

  // combine (m,l) across the 16 lanes sharing each output row
  float mR[8], lR[8];
#pragma unroll
  for (int r = 0; r < 8; ++r) {
    float m = mL[r], l = lL[r];
#pragma unroll
    for (int off = 1; off < 16; off <<= 1) {
      const float m2 = __shfl_xor(m, off, 32);
      const float l2 = __shfl_xor(l, off, 32);
      const float mx = fmaxf(m, m2);
      l = l * __expf(m - mx) + l2 * __expf(m2 - mx);
      m = mx;
    }
    mR[r] = m;
    lR[r] = 1.f / l;
  }

  // ---- pass 2: write normalized probs + accumulate O = P @ V ----
  v8f oacc[4];
#pragma unroll
  for (int nd = 0; nd < 4; ++nd)
#pragma unroll
    for (int r = 0; r < 8; ++r) oacc[nd][r] = 0.f;

  float* attn_base = attn + ((size_t)bh * S_ + q0) * S_;

  for (int k0 = 0; k0 < S_; k0 += KC_) {
    {  // async K chunk: 4 x b128 per lane straight into LDS (ASYNCcnt)
      const int row = t >> 1, c0 = (t & 1) * 32;
      const unsigned short* g = Kb + (size_t)(k0 + row) * D_ + c0;
      const unsigned l = lds_off(&Ks[row * D_ + c0]);
      async_b128(g, l);
      async_b128(g + 8, l + 16);
      async_b128(g + 16, l + 32);
      async_b128(g + 24, l + 48);
    }
    {  // V chunk staged transposed: thread owns token-pair (2p,2p+1), 16 d's.
      const int p = t >> 2, dg = (t & 3) * 16;
      const unsigned short* r0 = Vb + (size_t)(k0 + 2 * p) * D_ + dg;
      const uint4 a0 = ((const uint4*)r0)[0], a1 = ((const uint4*)r0)[1];
      const uint4 c0 = ((const uint4*)(r0 + D_))[0], c1 = ((const uint4*)(r0 + D_))[1];
      const unsigned ua[8] = {a0.x, a0.y, a0.z, a0.w, a1.x, a1.y, a1.z, a1.w};
      const unsigned ub[8] = {c0.x, c0.y, c0.z, c0.w, c1.x, c1.y, c1.z, c1.w};
#pragma unroll
      for (int j = 0; j < 8; ++j) {
        const unsigned lo = ua[j], hi = ub[j];
        *(unsigned int*)&Vt[(dg + 2 * j) * KC_ + 2 * p]     = (lo & 0xffffu) | (hi << 16);
        *(unsigned int*)&Vt[(dg + 2 * j + 1) * KC_ + 2 * p] = (lo >> 16) | (hi & 0xffff0000u);
      }
    }
    if (t < KC_) Ms[t] = mask[(size_t)b * S_ + k0 + t] * (-1e9f);
    if (k0 + KC_ < S_) {  // pull next chunk toward L2 (global_prefetch_b8)
      __builtin_prefetch(Kb + (size_t)(k0 + KC_) * D_ + t * 32, 0, 1);
      __builtin_prefetch(Vb + (size_t)(k0 + KC_) * D_ + t * 32, 0, 1);
    }
    wait_async0();
    __syncthreads();
#pragma unroll
    for (int kt2 = 0; kt2 < KC_ / 32; ++kt2) {
#pragma unroll
      for (int tt = 0; tt < 2; ++tt) {
        const int kt = kt2 * 2 + tt;
        v8f acc;
#pragma unroll
        for (int r = 0; r < 8; ++r) acc[r] = 0.f;
        acc = wmma_bf16(qa0, frag_b_cont(Ks, D_, kt * 16, 0, lane), acc);
        acc = wmma_bf16(qa1, frag_b_cont(Ks, D_, kt * 16, 32, lane), acc);
        const float mk = Ms[kt * 16 + nn];
#pragma unroll
        for (int r = 0; r < 8; ++r) {
          const float val = acc[r] * 0.125f + mk;
          const float p = __expf(val - mR[r]) * lR[r];
          const int R = r + 8 * half;
          attn_base[(size_t)R * S_ + k0 + kt * 16 + nn] = p;          // fp32 out
          Ps[wave][R * 32 + tt * 16 + nn] = f2bf(p);                  // for P@V
        }
      }
      // wave-local LDS RAW (DS ops in-order within a wave)
      const v16us pa = frag_a(&Ps[wave][0], 32, lane);
#pragma unroll
      for (int nd = 0; nd < 4; ++nd)
        oacc[nd] = wmma_bf16(pa, frag_b_cont(Vt, KC_, nd * 16, kt2 * 32, lane),
                             oacc[nd]);
    }
    __syncthreads();
  }

  // O tile (16 x 64) -> oh[b][s][h*64+d] (bf16, heads re-concatenated)
#pragma unroll
  for (int nd = 0; nd < 4; ++nd) {
    const int dd = nd * 16 + nn;
#pragma unroll
    for (int r = 0; r < 8; ++r) {
      const int s = q0 + r + 8 * half;
      oh[((size_t)b * S_ + s) * DM_ + h * D_ + dd] = f2bf(oacc[nd][r]);
    }
  }
}

// ---------------------------------------------------------------------------
// Kernel 3: out = O(bf16) * Wo(fp32->bf16) + bo, fp32 output [B*S][DM].
// A tile (bf16, strided rows) arrives via TDM; W tile staged transposed.
// ---------------------------------------------------------------------------
__global__ __launch_bounds__(256) void out_gemm_kernel(
    const unsigned short* __restrict__ X, const float* __restrict__ W,
    const float* __restrict__ bias, float* __restrict__ out) {
  __shared__ __align__(16) unsigned short As[128 * 32];
  __shared__ __align__(16) unsigned short Bt[64 * 32];
  const int t = threadIdx.x, lane = t & 31, wave = t >> 5;
  const int wm = wave >> 1, wn = wave & 1;
  const int m0 = blockIdx.y * 128, n0 = blockIdx.x * 64;

  v8f acc[2][2];
#pragma unroll
  for (int i = 0; i < 2; ++i)
#pragma unroll
    for (int j = 0; j < 2; ++j)
#pragma unroll
      for (int r = 0; r < 8; ++r) acc[i][j][r] = 0.f;

  for (int k0 = 0; k0 < DM_; k0 += 32) {
    if (t < 32)  // TDM: strided 128x32 bf16 tile (row stride DM_) -> As
      tdm_load_2d_bf16(X + (size_t)m0 * DM_ + k0, lds_off(As), 32, 128,
                       (unsigned)(NT_ - m0), DM_);
    {  // stage W tile 32x64 transposed (fp32 -> bf16 K-pair packing)
      const int pk = t >> 4, c = (t & 15) * 4;
      const float4 f0 = *(const float4*)(W + (size_t)(k0 + 2 * pk) * DM_ + n0 + c);
      const float4 f1 = *(const float4*)(W + (size_t)(k0 + 2 * pk + 1) * DM_ + n0 + c);
      *(unsigned int*)&Bt[(c + 0) * 32 + 2 * pk] = pack2(f0.x, f1.x);
      *(unsigned int*)&Bt[(c + 1) * 32 + 2 * pk] = pack2(f0.y, f1.y);
      *(unsigned int*)&Bt[(c + 2) * 32 + 2 * pk] = pack2(f0.z, f1.z);
      *(unsigned int*)&Bt[(c + 3) * 32 + 2 * pk] = pack2(f0.w, f1.w);
    }
    if (t < 32) wait_tensor0();
    __syncthreads();
    const v16us a0 = frag_a(&As[(wm * 32 + 0) * 32], 32, lane);
    const v16us a1 = frag_a(&As[(wm * 32 + 16) * 32], 32, lane);
    const v16us b0 = frag_b_cont(Bt, 32, wn * 32 + 0, 0, lane);
    const v16us b1 = frag_b_cont(Bt, 32, wn * 32 + 16, 0, lane);
    acc[0][0] = wmma_bf16(a0, b0, acc[0][0]);
    acc[0][1] = wmma_bf16(a0, b1, acc[0][1]);
    acc[1][0] = wmma_bf16(a1, b0, acc[1][0]);
    acc[1][1] = wmma_bf16(a1, b1, acc[1][1]);
    __syncthreads();
  }

  const int nn = lane & 15, half = (lane >> 4) & 1;
#pragma unroll
  for (int i = 0; i < 2; ++i)
#pragma unroll
    for (int j = 0; j < 2; ++j) {
      const int ncol = n0 + wn * 32 + j * 16 + nn;
      const float bv = bias[ncol];
#pragma unroll
      for (int r = 0; r < 8; ++r) {
        const int mrow = m0 + wm * 32 + i * 16 + r + 8 * half;
        out[(size_t)mrow * DM_ + ncol] = acc[i][j][r] + bv;
      }
    }
}

// ---------------------------------------------------------------------------
// Launch. Workspace: qh,kh,vh,oh bf16 buffers (4 * 4,194,304 ushorts = 32 MB).
// d_out = [out: 4096*1024 fp32][attn: 2*16*2048*2048 fp32].
// ---------------------------------------------------------------------------
extern "C" void kernel_launch(void* const* d_in, const int* in_sizes, int n_in,
                              void* d_out, int out_size, void* d_ws, size_t ws_size,
                              hipStream_t stream) {
  const float* q    = (const float*)d_in[0];
  const float* k    = (const float*)d_in[1];
  const float* v    = (const float*)d_in[2];
  const float* mask = (const float*)d_in[3];
  const float* Wq   = (const float*)d_in[4];
  const float* bq   = (const float*)d_in[5];
  const float* Wk   = (const float*)d_in[6];
  const float* bk   = (const float*)d_in[7];
  const float* Wv   = (const float*)d_in[8];
  const float* bv   = (const float*)d_in[9];
  const float* Wo   = (const float*)d_in[10];
  const float* bo   = (const float*)d_in[11];

  float* out  = (float*)d_out;
  float* attn = out + (size_t)NT_ * DM_;

  const size_t HSD = (size_t)B_ * H_ * S_ * D_;  // 4,194,304 elements
  unsigned short* qh = (unsigned short*)d_ws;
  unsigned short* kh = qh + HSD;
  unsigned short* vh = kh + HSD;
  unsigned short* oh = vh + HSD;                 // [B*S][DM] bf16

  const dim3 blk(256);
  const dim3 ggemm(DM_ / 64, NT_ / 128);         // (16, 32)
  const dim3 gattn(S_ / 128, B_ * H_);           // (16, 32)

  proj_gemm_kernel<<<ggemm, blk, 0, stream>>>(q, Wq, bq, qh);
  proj_gemm_kernel<<<ggemm, blk, 0, stream>>>(k, Wk, bk, kh);
  proj_gemm_kernel<<<ggemm, blk, 0, stream>>>(v, Wv, bv, vh);
  attn_kernel<<<gattn, blk, 0, stream>>>(qh, kh, vh, mask, attn, oh);
  out_gemm_kernel<<<ggemm, blk, 0, stream>>>(oh, Wo, bo, out);
}
